// SetDecoder_75488345194527
// MI455X (gfx1250) — matrix operand order repaired and checked by reading, not compiled, not run on hardware
//
#include <hip/hip_runtime.h>
#include <hip/hip_bf16.h>

// ---- problem constants (match reference) ----
#define Bc   4
#define Qc   32
#define Lc   512
#define Hc   768
#define NLc  3
#define NHc  12
#define FFc  3072
#define DHc  64
#define NCP1 25

typedef __attribute__((ext_vector_type(16))) _Float16 v16h;
typedef __attribute__((ext_vector_type(8)))  _Float16 v8h;
typedef __attribute__((ext_vector_type(8)))  float    v8f;

// =====================================================================
// WMMA GEMM: Y[M,N] = act(X[M,K] @ W[K,N] + bias[N])
// block = 128 threads (4 wave32), computes a 32x32 output tile,
// K-step 64 (two v_wmma_f32_16x16x32_f16 per stage).
// Fast path (NBOUND=false): requires M%32==0, N%32==0, K%64==0 -> pure
// global_load_b128 / ds_store_b128 / ds_load_b128 staging, no predication.
// NBOUND=true: guarded B loads / stores (used only for the N=25 cls GEMM).
// A tile in LDS row-major (k contiguous); B tile stored TRANSPOSED
// (n-major, k contiguous) so both WMMA fragments are two b128 LDS loads.
// =====================================================================
#define KT  64
#define LDA 72   // halfs per A row (144B, 16B aligned)
#define LDB 72   // halfs per Bt row

static __device__ inline v8h cvt8(const float* s) {
  float4 a = *reinterpret_cast<const float4*>(s);
  float4 b = *reinterpret_cast<const float4*>(s + 4);
  v8h h;
  h[0] = (_Float16)a.x; h[1] = (_Float16)a.y;
  h[2] = (_Float16)a.z; h[3] = (_Float16)a.w;
  h[4] = (_Float16)b.x; h[5] = (_Float16)b.y;
  h[6] = (_Float16)b.z; h[7] = (_Float16)b.w;
  return h;
}

static __device__ inline v16h concat8(v8h lo, v8h hi) {
  v16h r;
#pragma unroll
  for (int i = 0; i < 8; ++i) { r[i] = lo[i]; r[i + 8] = hi[i]; }
  return r;
}

template <bool NBOUND>
__global__ void wmma_gemm_kernel(const float* __restrict__ X,
                                 const float* __restrict__ W,
                                 const float* __restrict__ bias,
                                 float* __restrict__ Y,
                                 int M, int N, int K, int act) {
  __shared__ _Float16 As[32 * LDA];   // [m][k]   32 x 64
  __shared__ _Float16 Bt[32 * LDB];   // [n][k]   32 x 64 (transposed)
  const int tid  = threadIdx.x;
  const int lane = tid & 31;
  const int wave = tid >> 5;
  const int m0 = blockIdx.x * 32;
  const int n0 = blockIdx.y * 32;
  const int wm = (wave >> 1) * 16;
  const int wn = (wave & 1) * 16;

  // staging assignments
  const int ar  = tid >> 2;           // A row 0..31
  const int ac  = (tid & 3) * 16;     // A col seg {0,16,32,48}
  const int kr  = tid >> 1;           // B k-row 0..63
  const int nsb = (tid & 1) * 16;     // B n seg {0,16}

  // fragment gather bases (per-lane, documented CDNA5 WMMA layouts)
  const int arow = (wm + (lane & 15)) * LDA;
  const int akh  = (lane >> 4) * 8;   // A: lane-half selects K-block {0,8}
  const int brow = (wn + (lane & 15)) * LDB;
  const int bkh  = (lane >> 4) * 8;   // B: lane-half selects K-block {0,8}

  v8f c = {};

  for (int k0 = 0; k0 < K; k0 += KT) {
    // ---- stage A: 32x64 f32 -> f16, vector loads + b128 LDS stores ----
    {
      const float* src = X + (size_t)(m0 + ar) * K + (k0 + ac);
      *reinterpret_cast<v8h*>(&As[ar * LDA + ac])     = cvt8(src);
      *reinterpret_cast<v8h*>(&As[ar * LDA + ac + 8]) = cvt8(src + 8);
    }
    // ---- stage B transposed: 64x32 f32 -> Bt[n][k] ----
    if (!NBOUND) {
      const float* wsrc = W + (size_t)(k0 + kr) * N + (n0 + nsb);
#pragma unroll
      for (int j = 0; j < 16; j += 4) {
        float4 f = *reinterpret_cast<const float4*>(wsrc + j);
        Bt[(nsb + j + 0) * LDB + kr] = (_Float16)f.x;
        Bt[(nsb + j + 1) * LDB + kr] = (_Float16)f.y;
        Bt[(nsb + j + 2) * LDB + kr] = (_Float16)f.z;
        Bt[(nsb + j + 3) * LDB + kr] = (_Float16)f.w;
      }
    } else {
      const float* wrow = W + (size_t)(k0 + kr) * N;
#pragma unroll
      for (int j = 0; j < 16; ++j) {
        int gn = n0 + nsb + j;
        float f = (gn < N) ? wrow[gn] : 0.0f;
        Bt[(nsb + j) * LDB + kr] = (_Float16)f;
      }
    }
    // prefetch next K tile of X (global_prefetch_b8)
    if (k0 + KT < K) __builtin_prefetch(&X[(size_t)(m0 + ar) * K + k0 + KT + ac], 0, 1);
    __syncthreads();

    // ---- two WMMA steps over the 64-K stage ----
#pragma unroll
    for (int kb = 0; kb < KT; kb += 32) {
      v8h a0 = *reinterpret_cast<const v8h*>(&As[arow + kb + akh]);
      v8h a1 = *reinterpret_cast<const v8h*>(&As[arow + kb + akh + 16]);
      v8h b0 = *reinterpret_cast<const v8h*>(&Bt[brow + kb + bkh]);
      v8h b1 = *reinterpret_cast<const v8h*>(&Bt[brow + kb + bkh + 16]);
      v16h a = concat8(a0, a1);
      v16h b = concat8(b0, b1);
      c = __builtin_amdgcn_wmma_f32_16x16x32_f16(
          /*neg_a=*/false, a, /*neg_b=*/false, b,
          /*c_mod=*/(short)0, c, /*reuse_a=*/false, /*reuse_b=*/false);
    }
    __syncthreads();
  }

  // ---- epilogue: C layout: VGPR r -> row (r + 8*(lane>>4)), col lane&15 ----
  const int n = n0 + wn + (lane & 15);
  if (!NBOUND || n < N) {
    const float bv = bias[n];
#pragma unroll
    for (int r = 0; r < 8; ++r) {
      int m = m0 + wm + r + (lane >> 4) * 8;
      float v = c[r] + bv;
      if (act == 1) { // gelu, tanh approximation (jax.nn.gelu default)
        float x3 = v * v * v;
        v = 0.5f * v * (1.0f + tanhf(0.7978845608028654f * (v + 0.044715f * x3)));
      }
      Y[(size_t)m * N + n] = v;
    }
  }
}

// =====================================================================
// Fused residual-add + LayerNorm: Y[r,:] = LN(X[r,:] + R[r % rmod,:])
// =====================================================================
__global__ void add_ln_kernel(const float* __restrict__ X,
                              const float* __restrict__ R, int rmod,
                              const float* __restrict__ g,
                              const float* __restrict__ bt,
                              float* __restrict__ Y) {
  __shared__ float row[Hc];
  __shared__ float red[8];
  const int r   = blockIdx.x;
  const int tid = threadIdx.x;
  const float* x  = X + (size_t)r * Hc;
  const float* rs = R + (size_t)(r % rmod) * Hc;

  float ls = 0.0f;
  for (int i = tid; i < Hc; i += 256) {
    float v = x[i] + rs[i];
    row[i] = v;
    ls += v;
  }
  for (int o = 16; o; o >>= 1) ls += __shfl_xor(ls, o);
  if ((tid & 31) == 0) red[tid >> 5] = ls;
  __syncthreads();
  float mean = 0.0f;
  for (int w = 0; w < 8; ++w) mean += red[w];
  mean *= (1.0f / (float)Hc);
  __syncthreads();

  float lv = 0.0f;
  for (int i = tid; i < Hc; i += 256) {
    float d = row[i] - mean;
    lv += d * d;
  }
  for (int o = 16; o; o >>= 1) lv += __shfl_xor(lv, o);
  if ((tid & 31) == 0) red[tid >> 5] = lv;
  __syncthreads();
  float var = 0.0f;
  for (int w = 0; w < 8; ++w) var += red[w];
  var *= (1.0f / (float)Hc);
  const float inv = rsqrtf(var + 1e-12f);

  for (int i = tid; i < Hc; i += 256)
    Y[(size_t)r * Hc + i] = (row[i] - mean) * inv * g[i] + bt[i];
}

// =====================================================================
// Self-attention (keys = Q = 32): one wave32 per (b, head, q)
// =====================================================================
__global__ void self_attn_kernel(const float* __restrict__ Qb,
                                 const float* __restrict__ Kb,
                                 const float* __restrict__ Vb,
                                 float* __restrict__ O) {
  __shared__ float as[Qc];
  const int lane = threadIdx.x;            // 0..31 == key index
  const int idx  = blockIdx.x;             // b*NH*Q + head*Q + q
  const int q    = idx % Qc;
  const int t    = idx / Qc;
  const int head = t % NHc;
  const int b    = t / NHc;

  const float* qrow = Qb + ((size_t)(b * Qc + q)) * Hc + head * DHc;
  const float* krow = Kb + ((size_t)(b * Qc + lane)) * Hc + head * DHc;
  float s = 0.0f;
#pragma unroll 8
  for (int d = 0; d < DHc; ++d) s += qrow[d] * krow[d];
  s *= 0.125f;  // 1/sqrt(64)

  float m = s;
  for (int o = 16; o; o >>= 1) m = fmaxf(m, __shfl_xor(m, o));
  float e = expf(s - m);
  float sum = e;
  for (int o = 16; o; o >>= 1) sum += __shfl_xor(sum, o);
  as[lane] = e / sum;
  __syncthreads();

  float o0 = 0.0f, o1 = 0.0f;
  for (int k = 0; k < Qc; ++k) {
    const float* vrow = Vb + ((size_t)(b * Qc + k)) * Hc + head * DHc;
    float ak = as[k];
    o0 += ak * vrow[lane];
    o1 += ak * vrow[lane + 32];
  }
  float* orow = O + ((size_t)(b * Qc + q)) * Hc + head * DHc;
  orow[lane]      = o0;
  orow[lane + 32] = o1;
}

// =====================================================================
// Cross-attention (keys = L = 512), masked. 256 threads per (b, head, q)
// =====================================================================
__global__ void cross_attn_kernel(const float* __restrict__ Qb,
                                  const float* __restrict__ Kb,
                                  const float* __restrict__ Vb,
                                  const int* __restrict__ mask,
                                  float* __restrict__ O) {
  __shared__ float qv[DHc];
  __shared__ float sc[Lc];
  __shared__ float red[8];
  __shared__ float oacc[4][DHc];
  const int tid  = threadIdx.x;            // 0..255
  const int idx  = blockIdx.x;
  const int q    = idx % Qc;
  const int t    = idx / Qc;
  const int head = t % NHc;
  const int b    = t / NHc;

  if (tid < DHc) qv[tid] = Qb[((size_t)(b * Qc + q)) * Hc + head * DHc + tid];
  __syncthreads();

  float lmax = -3.0e38f;
  for (int k = tid; k < Lc; k += 256) {
    const float* krow = Kb + ((size_t)(b * Lc + k)) * Hc + head * DHc;
    float s = 0.0f;
#pragma unroll 8
    for (int d = 0; d < DHc; ++d) s += qv[d] * krow[d];
    s = s * 0.125f + (mask[b * Lc + k] ? 0.0f : -10000.0f);
    sc[k] = s;
    lmax = fmaxf(lmax, s);
  }
  for (int o = 16; o; o >>= 1) lmax = fmaxf(lmax, __shfl_xor(lmax, o));
  if ((tid & 31) == 0) red[tid >> 5] = lmax;
  __syncthreads();
  float bmax = red[0];
  for (int w = 1; w < 8; ++w) bmax = fmaxf(bmax, red[w]);
  __syncthreads();

  float lsum = 0.0f;
  for (int k = tid; k < Lc; k += 256) {
    float e = expf(sc[k] - bmax);
    sc[k] = e;
    lsum += e;
  }
  for (int o = 16; o; o >>= 1) lsum += __shfl_xor(lsum, o);
  if ((tid & 31) == 0) red[tid >> 5] = lsum;
  __syncthreads();
  float bsum = 0.0f;
  for (int w = 0; w < 8; ++w) bsum += red[w];
  const float inv = 1.0f / bsum;
  __syncthreads();

  const int d  = tid & 63;
  const int kg = tid >> 6;
  float acc = 0.0f;
  for (int k = kg * 128; k < kg * 128 + 128; ++k)
    acc += sc[k] * Vb[((size_t)(b * Lc + k)) * Hc + head * DHc + d];
  oacc[kg][d] = acc;
  __syncthreads();
  if (tid < DHc) {
    float o = (oacc[0][tid] + oacc[1][tid] + oacc[2][tid] + oacc[3][tid]) * inv;
    O[((size_t)(b * Qc + q)) * Hc + head * DHc + tid] = o;
  }
}

// =====================================================================
// Biaffine head: out[b,q,l] = sum_h tanh(HQ[b,q,h] + HE[b,l,h]) * w3[h]
// =====================================================================
__global__ void biaffine_kernel(const float* __restrict__ HQ,
                                const float* __restrict__ HE,
                                const float* __restrict__ w3,
                                float* __restrict__ out) {
  __shared__ float hq[Hc];
  __shared__ float wv[Hc];
  const int r   = blockIdx.x;   // b*Q + q
  const int b   = r >> 5;       // Q == 32
  const int tid = threadIdx.x;
  for (int i = tid; i < Hc; i += 256) {
    hq[i] = HQ[(size_t)r * Hc + i];
    wv[i] = w3[i];
  }
  __syncthreads();
  for (int l = tid; l < Lc; l += 256) {
    const float* he = HE + ((size_t)(b * Lc + l)) * Hc;
    float s = 0.0f;
    for (int h = 0; h < Hc; ++h) s += tanhf(hq[h] + he[h]) * wv[h];
    out[(size_t)r * Lc + l] = s;
  }
}

// =====================================================================
// host-side orchestration
// =====================================================================
static inline void gemm(const float* X, const float* W, const float* bias,
                        float* Y, int M, int N, int K, int act, hipStream_t s) {
  dim3 g((M + 31) / 32, (N + 31) / 32);
  if ((N & 31) == 0)
    wmma_gemm_kernel<false><<<g, 128, 0, s>>>(X, W, bias, Y, M, N, K, act);
  else
    wmma_gemm_kernel<true><<<g, 128, 0, s>>>(X, W, bias, Y, M, N, K, act);
}

extern "C" void kernel_launch(void* const* d_in, const int* in_sizes, int n_in,
                              void* d_out, int out_size, void* d_ws, size_t ws_size,
                              hipStream_t stream) {
  (void)in_sizes; (void)n_in; (void)out_size; (void)ws_size;

  const float* hidden = (const float*)d_in[0];
  const float* enc    = (const float*)d_in[1];
  const int*   mask   = (const int*)d_in[2];
  const float* qemb   = (const float*)d_in[3];
  const float* ln0g   = (const float*)d_in[4];
  const float* ln0b   = (const float*)d_in[5];
  // 6..13  sa_{q,k,v,o}{w,b}, 14..21 ca_{q,k,v,o}{w,b}
  // 22..27 ln{1,2,3}_{g,b}; 28..31 ff_{w1,b1,w2,b2}; 32,33 cls_{w,b}
  // 34.. 4x biaffine heads (w1,b1,w2,b2,w3)

  float* ws = (float*)d_ws;
  float* h   = ws;                 // [B*Q, H]      98304
  float* t0  = ws +   98304;       // [B*Q, H]
  float* t1  = ws +  196608;       // [B*Q, H]
  float* t2  = ws +  294912;       // [B*Q, H]
  float* t3  = ws +  393216;       // [B*Q, H]
  float* t4  = ws +  491520;       // [B*Q, H]
  float* ff  = ws +  589824;       // [B*Q, FF]     393216
  float* kc  = ws +  983040;       // [B*L, H]      1572864
  float* vc  = ws + 2555904;       // [B*L, H]      1572864
  float* heb = ws + 4128768;       // [B*L, H]      1572864

  const int M1 = Bc * Qc;   // 128
  const int M2 = Bc * Lc;   // 2048
  const size_t WW = (size_t)Hc * Hc;

  // h = LN(hidden + query_embed[broadcast over B])
  add_ln_kernel<<<M1, 256, 0, stream>>>(hidden, qemb, Qc, ln0g, ln0b, h);

  for (int i = 0; i < NLc; ++i) {
    const float* saqw = (const float*)d_in[6]  + (size_t)i * WW;
    const float* saqb = (const float*)d_in[7]  + (size_t)i * Hc;
    const float* sakw = (const float*)d_in[8]  + (size_t)i * WW;
    const float* sakb = (const float*)d_in[9]  + (size_t)i * Hc;
    const float* savw = (const float*)d_in[10] + (size_t)i * WW;
    const float* savb = (const float*)d_in[11] + (size_t)i * Hc;
    const float* saow = (const float*)d_in[12] + (size_t)i * WW;
    const float* saob = (const float*)d_in[13] + (size_t)i * Hc;
    const float* caqw = (const float*)d_in[14] + (size_t)i * WW;
    const float* caqb = (const float*)d_in[15] + (size_t)i * Hc;
    const float* cakw = (const float*)d_in[16] + (size_t)i * WW;
    const float* cakb = (const float*)d_in[17] + (size_t)i * Hc;
    const float* cavw = (const float*)d_in[18] + (size_t)i * WW;
    const float* cavb = (const float*)d_in[19] + (size_t)i * Hc;
    const float* caow = (const float*)d_in[20] + (size_t)i * WW;
    const float* caob = (const float*)d_in[21] + (size_t)i * Hc;
    const float* ln1g = (const float*)d_in[22] + (size_t)i * Hc;
    const float* ln1b = (const float*)d_in[23] + (size_t)i * Hc;
    const float* ln2g = (const float*)d_in[24] + (size_t)i * Hc;
    const float* ln2b = (const float*)d_in[25] + (size_t)i * Hc;
    const float* ln3g = (const float*)d_in[26] + (size_t)i * Hc;
    const float* ln3b = (const float*)d_in[27] + (size_t)i * Hc;
    const float* ffw1 = (const float*)d_in[28] + (size_t)i * Hc * FFc;
    const float* ffb1 = (const float*)d_in[29] + (size_t)i * FFc;
    const float* ffw2 = (const float*)d_in[30] + (size_t)i * FFc * Hc;
    const float* ffb2 = (const float*)d_in[31] + (size_t)i * Hc;

    // ---- self attention ----
    gemm(h, saqw, saqb, t0, M1, Hc, Hc, 0, stream);
    gemm(h, sakw, sakb, t1, M1, Hc, Hc, 0, stream);
    gemm(h, savw, savb, t2, M1, Hc, Hc, 0, stream);
    self_attn_kernel<<<Bc * NHc * Qc, 32, 0, stream>>>(t0, t1, t2, t3);
    gemm(t3, saow, saob, t4, M1, Hc, Hc, 0, stream);
    add_ln_kernel<<<M1, 256, 0, stream>>>(t4, h, M1, ln1g, ln1b, h);

    // ---- cross attention ----
    gemm(h,   caqw, caqb, t0, M1, Hc, Hc, 0, stream);
    gemm(enc, cakw, cakb, kc, M2, Hc, Hc, 0, stream);
    gemm(enc, cavw, cavb, vc, M2, Hc, Hc, 0, stream);
    cross_attn_kernel<<<Bc * NHc * Qc, 256, 0, stream>>>(t0, kc, vc, mask, t3);
    gemm(t3, caow, caob, t4, M1, Hc, Hc, 0, stream);
    add_ln_kernel<<<M1, 256, 0, stream>>>(t4, h, M1, ln2g, ln2b, h);

    // ---- FFN ----
    gemm(h,  ffw1, ffb1, ff, M1, FFc, Hc, 1, stream);   // fused GELU
    gemm(ff, ffw2, ffb2, t4, M1, Hc, FFc, 0, stream);
    add_ln_kernel<<<M1, 256, 0, stream>>>(t4, h, M1, ln3g, ln3b, h);
  }

  float* out = (float*)d_out;
  // class_logits [B,Q,NC+1]  (N=25 -> bounded path)
  gemm(h, (const float*)d_in[32], (const float*)d_in[33], out,
       M1, NCP1, Hc, 0, stream);

  // 4 biaffine heads: hs, he, ts, te -> [B,Q,L] each
  const int   hidx[4]  = {34, 39, 44, 49};
  float*      houts[4] = {out + 3200, out + 68736, out + 134272, out + 199808};
  for (int p = 0; p < 4; ++p) {
    const float* w1 = (const float*)d_in[hidx[p] + 0];
    const float* b1 = (const float*)d_in[hidx[p] + 1];
    const float* w2 = (const float*)d_in[hidx[p] + 2];
    const float* b2 = (const float*)d_in[hidx[p] + 3];
    const float* w3 = (const float*)d_in[hidx[p] + 4];
    gemm(h,   w1, b1, t0,  M1, Hc, Hc, 0, stream);
    gemm(enc, w2, b2, heb, M2, Hc, Hc, 0, stream);
    biaffine_kernel<<<M1, 256, 0, stream>>>(t0, heb, w3, houts[p]);
  }
}
